// GraphSAGEReasoner_70368744178309
// MI455X (gfx1250) — matrix operand order repaired and checked by reading, not compiled, not run on hardware
//
#include <hip/hip_runtime.h>

// ---------------------------------------------------------------------------
// GraphSAGE + LSTM + classifier for MI455X (gfx1250), bf16 WMMA f32-accum.
// B=256, L=4, N=100000, EMB=128, NEI=32, SFW=384.
// GEMM: 32x(NT*16) tile per wave; per k-step one grouped load burst feeds
// 2*NT back-to-back v_wmma_f32_16x16x32_bf16. Single fragment buffer keeps
// VGPR pressure ~130 (no scratch spills, unlike double-buffered variant).
// ---------------------------------------------------------------------------

#define DIV_UP(a, b) (((a) + (b) - 1) / (b))

typedef __attribute__((ext_vector_type(16))) __bf16 v16bf;
typedef __attribute__((ext_vector_type(8)))  float  v8f;

union V16 {
  v16bf v;
  uint4 q[2];
};

__device__ __forceinline__ unsigned short f2bf(float f) {
  unsigned u = __float_as_uint(f);
  u += 0x7FFFu + ((u >> 16) & 1u);            // round-to-nearest-even
  return (unsigned short)(u >> 16);
}

// ---- f32 -> bf16 weight conversion ----------------------------------------
__global__ void convert_f32_to_bf16(const float* __restrict__ src,
                                    unsigned short* __restrict__ dst, int n) {
  int i = blockIdx.x * blockDim.x + threadIdx.x;
  if (i < n) dst[i] = f2bf(src[i]);
}

// ---- 2-hop gather + leaf mean; builds X1 = [emb[n1] | mean(emb[n2])] bf16 --
// One wave per output row r in [0, 32768): bl = r>>5, j = r&31.
__global__ void gather_agg2_kernel(const int* __restrict__ path,
                                   const int* __restrict__ neigh,
                                   const float* __restrict__ emb,
                                   unsigned short* __restrict__ X1bf) {
  int w    = (blockIdx.x * blockDim.x + threadIdx.x) >> 5;
  int lane = threadIdx.x & 31;
  if (w >= 1024 * 32) return;
  int bl = w >> 5, j = w & 31;
  int pid = path[bl];
  int e   = neigh[pid * 32 + j];

  float4 s = ((const float4*)(emb + (long)e * 128))[lane];

  int leaves[16];
#pragma unroll
  for (int k = 0; k < 16; ++k) leaves[k] = neigh[e * 32 + k];

  float ax = 0.f, ay = 0.f, az = 0.f, aw = 0.f;
#pragma unroll
  for (int k = 0; k < 16; ++k) {
    if (k + 1 < 16)   // hide L2 gather latency: global_prefetch next leaf row
      __builtin_prefetch((const char*)(emb + (long)leaves[k + 1] * 128) + lane * 16, 0, 0);
    float4 v = ((const float4*)(emb + (long)leaves[k] * 128))[lane];
    ax += v.x; ay += v.y; az += v.z; aw += v.w;
  }
  const float inv = 1.0f / 16.0f;
  unsigned short* dst = X1bf + (long)w * 256;
  uint2 us, ua;
  us.x = (unsigned)f2bf(s.x)      | ((unsigned)f2bf(s.y)      << 16);
  us.y = (unsigned)f2bf(s.z)      | ((unsigned)f2bf(s.w)      << 16);
  ua.x = (unsigned)f2bf(ax * inv) | ((unsigned)f2bf(ay * inv) << 16);
  ua.y = (unsigned)f2bf(az * inv) | ((unsigned)f2bf(aw * inv) << 16);
  ((uint2*)dst)[lane]         = us;   // cols [0,128)  : emb[n1]
  ((uint2*)(dst + 128))[lane] = ua;   // cols [128,256): agg2
}

// ---- Generic bf16 WMMA GEMM: C = act(A@B + bias + addend) ------------------
// A row-major [M,K] bf16, B row-major [K,N] bf16, K = 32*kSteps.
// Wave tile: (MT*16) rows x (NT*16) cols; fragments live only per k-step.
template <int MT, int NT>
__global__ void gemm_bf16_wmma(const unsigned short* __restrict__ A, int lda,
                               const unsigned short* __restrict__ B, int ldb,
                               const float* __restrict__ bias,
                               const float* __restrict__ addend, int ldadd,
                               float* __restrict__ Cf, int ldc,
                               unsigned short* __restrict__ Cb, int ldcb,
                               int mTiles, int nStrips, int kSteps, int doRelu) {
  int w    = (blockIdx.x * blockDim.x + threadIdx.x) >> 5;
  int lane = threadIdx.x & 31;
  if (w >= mTiles * nStrips) return;            // whole-wave exit, EXEC stays all-1s
  int mt = w / nStrips, ns = w % nStrips;
  int m0 = mt * (MT * 16), ncol0 = ns * (NT * 16);
  int m = lane & 15, half = lane >> 4;
  int kb = half * 8;                            // first B K-row of this lane (table order)

  const unsigned short* ap[MT];
#pragma unroll
  for (int mi = 0; mi < MT; ++mi)
    ap[mi] = A + (long)(m0 + mi * 16 + m) * lda + half * 8;
  const unsigned short* bp = B + (long)kb * ldb + ncol0 + m;
  const long bstep = 32L * ldb;

  v8f acc[MT][NT] = {};

  for (int ks = 0; ks < kSteps; ++ks) {
    V16 a[MT], b[NT];
#pragma unroll
    for (int mi = 0; mi < MT; ++mi) {           // grouped load burst (s_clause)
      a[mi].q[0] = *(const uint4*)(ap[mi]);     // K sub 0..7
      a[mi].q[1] = *(const uint4*)(ap[mi] + 16);// K sub 16..23
      ap[mi] += 32;
    }
#pragma unroll
    for (int jj = 0; jj < NT; ++jj) {
#pragma unroll
      for (int s = 0; s < 16; ++s)
        b[jj].v[s] = __builtin_bit_cast(__bf16, bp[jj * 16 + (long)(s < 8 ? s : s + 8) * ldb]);
    }
    bp += bstep;
#pragma unroll
    for (int mi = 0; mi < MT; ++mi)             // 2*NT back-to-back WMMAs
#pragma unroll
      for (int jj = 0; jj < NT; ++jj)
        acc[mi][jj] = __builtin_amdgcn_wmma_f32_16x16x32_bf16(
            false, a[mi].v, false, b[jj].v, (short)0, acc[mi][jj], false, false);
  }

#pragma unroll
  for (int mi = 0; mi < MT; ++mi) {
    int mb = m0 + mi * 16 + half * 8;           // C/D layout: lanes16-31 -> M+8
#pragma unroll
    for (int jj = 0; jj < NT; ++jj) {
      int n = ncol0 + jj * 16 + m;
      float bv = bias ? bias[n] : 0.0f;
#pragma unroll
      for (int v = 0; v < 8; ++v) {
        float x = acc[mi][jj][v] + bv;
        if (addend) x += addend[(long)(mb + v) * ldadd + n];
        if (doRelu) x = fmaxf(x, 0.0f);
        if (Cf) Cf[(long)(mb + v) * ldc  + n] = x;
        if (Cb) Cb[(long)(mb + v) * ldcb + n] = f2bf(x);
      }
    }
  }
}

// ---- agg1 = mean_j f1[bl*32+j,:]; X0 = [emb[path] | agg1] bf16 -------------
__global__ void agg1_x0_kernel(const float* __restrict__ f1,
                               const int* __restrict__ path,
                               const float* __restrict__ emb,
                               unsigned short* __restrict__ X0bf) {
  int bl = blockIdx.x;        // 0..1023
  int t  = threadIdx.x;       // 0..255
  float s = 0.f;
  const float* base = f1 + (long)bl * 32 * 256 + t;
#pragma unroll
  for (int j = 0; j < 32; ++j) s += base[j * 256];
  X0bf[(long)bl * 384 + 128 + t] = f2bf(s * (1.0f / 32.0f));
  if (t < 128) {
    int pid = path[bl];
    X0bf[(long)bl * 384 + t] = f2bf(emb[(long)pid * 128 + t]);
  }
}

// ---- LSTM gates (Keras order i,f,g,o): updates c (f32) and h (bf16) --------
__global__ void lstm_gate_kernel(const float* __restrict__ Z,
                                 float* __restrict__ c,
                                 unsigned short* __restrict__ hbf) {
  int idx = blockIdx.x * blockDim.x + threadIdx.x;
  if (idx >= 256 * 384) return;
  int b = idx / 384, n = idx % 384;
  const float* z = Z + (long)b * 1536;
  float zi = z[n], zf = z[384 + n], zg = z[768 + n], zo = z[1152 + n];
  float si = 1.0f / (1.0f + __expf(-zi));
  float sf = 1.0f / (1.0f + __expf(-zf));
  float so = 1.0f / (1.0f + __expf(-zo));
  float g  = tanhf(zg);
  float cv = sf * c[idx] + si * g;
  c[idx]   = cv;
  hbf[idx] = f2bf(so * tanhf(cv));
}

// ---- final [400x2] projection + softmax: one wave per row ------------------
__global__ void classifier_out_kernel(const float* __restrict__ hid2,
                                      const float* __restrict__ Wc3,
                                      const float* __restrict__ bc3,
                                      float* __restrict__ out) {
  int w    = (blockIdx.x * blockDim.x + threadIdx.x) >> 5;
  int lane = threadIdx.x & 31;
  if (w >= 256) return;
  float s0 = 0.f, s1 = 0.f;
  for (int k = lane; k < 400; k += 32) {
    float hv = hid2[(long)w * 400 + k];
    s0 += hv * Wc3[k * 2 + 0];
    s1 += hv * Wc3[k * 2 + 1];
  }
#pragma unroll
  for (int off = 16; off > 0; off >>= 1) {
    s0 += __shfl_xor(s0, off, 32);
    s1 += __shfl_xor(s1, off, 32);
  }
  if (lane == 0) {
    s0 += bc3[0]; s1 += bc3[1];
    float mx = fmaxf(s0, s1);
    float e0 = __expf(s0 - mx), e1 = __expf(s1 - mx);
    float inv = 1.0f / (e0 + e1);
    out[w * 2 + 0] = e0 * inv;
    out[w * 2 + 1] = e1 * inv;
  }
}

// ---------------------------------------------------------------------------
extern "C" void kernel_launch(void* const* d_in, const int* in_sizes, int n_in,
                              void* d_out, int out_size, void* d_ws, size_t ws_size,
                              hipStream_t stream) {
  (void)in_sizes; (void)n_in; (void)out_size; (void)ws_size;
  const int*   path = (const int*)d_in[0];     // [256*4]
  const int*   neigh = (const int*)d_in[1];    // [100000*32]
  const float* emb  = (const float*)d_in[2];   // [100000*128]
  const float* W0   = (const float*)d_in[3];   // [384*384]
  const float* b0   = (const float*)d_in[4];
  const float* W1   = (const float*)d_in[5];   // [256*256]
  const float* b1   = (const float*)d_in[6];
  const float* Klstm = (const float*)d_in[7];  // [384*1536]
  const float* Rlstm = (const float*)d_in[8];  // [384*1536]
  const float* lb   = (const float*)d_in[9];   // [1536]
  const float* Wc1  = (const float*)d_in[10];  // [384*400]
  const float* bc1  = (const float*)d_in[11];
  const float* Wc2  = (const float*)d_in[12];  // [400*400]
  const float* bc2  = (const float*)d_in[13];
  const float* Wc3  = (const float*)d_in[14];  // [400*2]
  const float* bc3  = (const float*)d_in[15];
  float* out = (float*)d_out;                  // [256*2] f32

  // ---- workspace carve-out (256B aligned) ----
  char* ws = (char*)d_ws;
  size_t off = 0;
  auto alloc = [&](size_t bytes) -> void* {
    void* p = ws + off;
    off = (off + bytes + 255) & ~(size_t)255;
    return p;
  };
  unsigned short* W1bf   = (unsigned short*)alloc((size_t)256 * 256 * 2);
  unsigned short* W0bf   = (unsigned short*)alloc((size_t)384 * 384 * 2);
  unsigned short* Kbf    = (unsigned short*)alloc((size_t)384 * 1536 * 2);
  unsigned short* Rbf    = (unsigned short*)alloc((size_t)384 * 1536 * 2);
  unsigned short* Wc1bf  = (unsigned short*)alloc((size_t)384 * 400 * 2);
  unsigned short* Wc2bf  = (unsigned short*)alloc((size_t)448 * 400 * 2); // K-pad->448
  unsigned short* X1bf   = (unsigned short*)alloc((size_t)32768 * 256 * 2);
  float*          f1f    = (float*)alloc((size_t)32768 * 256 * 4);
  unsigned short* X0bf   = (unsigned short*)alloc((size_t)1024 * 384 * 2);
  unsigned short* f0bf   = (unsigned short*)alloc((size_t)1024 * 384 * 2);
  float*          ZX     = (float*)alloc((size_t)1024 * 1536 * 4);
  float*          Zt     = (float*)alloc((size_t)256 * 1536 * 4);
  unsigned short* hbf    = (unsigned short*)alloc((size_t)256 * 384 * 2);
  float*          cst    = (float*)alloc((size_t)256 * 384 * 4);
  unsigned short* hid1bf = (unsigned short*)alloc((size_t)256 * 448 * 2); // K-pad->448
  float*          hid2f  = (float*)alloc((size_t)256 * 400 * 4);

  // zero state / pad regions every call (deterministic, graph-capturable)
  hipMemsetAsync(Wc2bf,  0, (size_t)448 * 400 * 2, stream);
  hipMemsetAsync(hbf,    0, (size_t)256 * 384 * 2, stream);
  hipMemsetAsync(cst,    0, (size_t)256 * 384 * 4, stream);
  hipMemsetAsync(hid1bf, 0, (size_t)256 * 448 * 2, stream);

  auto cvt = [&](const float* s, unsigned short* d, int n) {
    convert_f32_to_bf16<<<DIV_UP(n, 256), 256, 0, stream>>>(s, d, n);
  };
  cvt(W1, W1bf, 256 * 256);
  cvt(W0, W0bf, 384 * 384);
  cvt(Klstm, Kbf, 384 * 1536);
  cvt(Rlstm, Rbf, 384 * 1536);
  cvt(Wc1, Wc1bf, 384 * 400);
  cvt(Wc2, Wc2bf, 400 * 400);   // pad rows 400..447 stay zero from memset

  // 2-hop gather + leaf mean -> X1 bf16 [32768, 256]   (L2-resident emb table)
  gather_agg2_kernel<<<4096, 256, 0, stream>>>(path, neigh, emb, X1bf);

  // f1 = relu(X1 @ W1 + b1) -> f32 [32768, 256]
  gemm_bf16_wmma<2, 4><<<512, 256, 0, stream>>>(
      X1bf, 256, W1bf, 256, b1, nullptr, 0, f1f, 256, nullptr, 0,
      /*mTiles=*/1024, /*nStrips=*/4, /*kSteps=*/8, /*relu=*/1);

  // agg1 mean over 32 neighbors; X0 = [emb[path] | agg1] bf16 [1024, 384]
  agg1_x0_kernel<<<1024, 256, 0, stream>>>(f1f, path, emb, X0bf);

  // f0 = relu(X0 @ W0 + b0) -> bf16 [1024, 384]
  gemm_bf16_wmma<2, 4><<<24, 256, 0, stream>>>(
      X0bf, 384, W0bf, 384, b0, nullptr, 0, nullptr, 0, f0bf, 384,
      /*mTiles=*/32, /*nStrips=*/6, /*kSteps=*/12, /*relu=*/1);

  // ZX = f0 @ lstm_kernel + lstm_bias -> f32 [1024, 1536] (hoisted out of scan)
  gemm_bf16_wmma<2, 4><<<96, 256, 0, stream>>>(
      f0bf, 384, Kbf, 1536, lb, nullptr, 0, ZX, 1536, nullptr, 0,
      /*mTiles=*/32, /*nStrips=*/24, /*kSteps=*/12, /*relu=*/0);

  // LSTM scan over L=4: Z = ZX[:,t,:] + h @ R; gates update (c, h)
  for (int t = 0; t < 4; ++t) {
    gemm_bf16_wmma<2, 4><<<24, 256, 0, stream>>>(
        hbf, 384, Rbf, 1536, nullptr, /*addend=*/ZX + t * 1536, /*ldadd=*/6144,
        Zt, 1536, nullptr, 0,
        /*mTiles=*/8, /*nStrips=*/24, /*kSteps=*/12, /*relu=*/0);
    lstm_gate_kernel<<<384, 256, 0, stream>>>(Zt, cst, hbf);
  }

  // classifier: Dense(400,relu) -> Dense(400,relu) -> Dense(2) + softmax
  gemm_bf16_wmma<2, 1><<<25, 256, 0, stream>>>(
      hbf, 384, Wc1bf, 400, bc1, nullptr, 0, nullptr, 0, hid1bf, 448,
      /*mTiles=*/8, /*nStrips=*/25, /*kSteps=*/12, /*relu=*/1);
  gemm_bf16_wmma<2, 1><<<25, 256, 0, stream>>>(
      hid1bf, 448, Wc2bf, 400, bc2, nullptr, 0, hid2f, 400, nullptr, 0,
      /*mTiles=*/8, /*nStrips=*/25, /*kSteps=*/14, /*relu=*/1);
  classifier_out_kernel<<<32, 256, 0, stream>>>(hid2f, Wc3, bc3, out);
}
